// GLABlockCLS_84748294684694
// MI455X (gfx1250) — compile-verified, matrix operand order checked by reading, not run on hardware
//
#include <hip/hip_runtime.h>
#include <hip/hip_bf16.h>
#include <math.h>

// ---------------- constants (from reference) ----------------
#define BB 8
#define SS 2048
#define DD 1024
#define HH 4
#define KK 512          // key dim
#define VV 1024         // value dim
#define DKk 128
#define DVv 256
#define TT (BB*SS)      // 16384 tokens
#define MLPH 4096
#define EPSF 1e-5f
#define GATE_NORM 16.0f

typedef __attribute__((ext_vector_type(16))) __bf16 bfx16;
typedef __attribute__((ext_vector_type(8)))  __bf16 bfx8;
typedef __attribute__((ext_vector_type(8)))  float  v8f;

union FragU { bfx16 v; bfx8 h[2]; };

// ---------------- WMMA fragment loader ----------------------
// 16-bit A/B fragment for v_wmma_f32_16x16x32_bf16.
// Lanes 0-15: row = row0+lane, K in {k0..k0+7} U {k0+16..k0+23}
// Lanes16-31: row = row0+lane-16, K in {k0+8..k0+15} U {k0+24..k0+31}
__device__ inline bfx16 load_frag(const __bf16* __restrict__ base, int row0,
                                  int ld, int k0, int lane) {
    int r  = row0 + (lane & 15);
    int ks = k0 + ((lane >> 4) ? 8 : 0);
    const __bf16* p = base + (size_t)r * ld + ks;
    FragU f;
    f.h[0] = *(const bfx8*)(p);
    f.h[1] = *(const bfx8*)(p + 16);
    return f.v;
}

__device__ inline float sigmoidf_(float x) { return 1.0f / (1.0f + __expf(-x)); }
__device__ inline float gelu_exact(float x) {
    return 0.5f * x * (1.0f + erff(x * 0.70710678118654752f));
}

// ---------------- generic bf16 WMMA GEMM --------------------
// C[M,N] = A[M,Kd] (bf16, row-major) @ Bt[N,Kd]^T (bf16, row-major = W^T)
// Wave tile 32x64 (2x4 WMMA accumulators); block = 8 waves -> 64x256 tile.
// epi: 0 -> Cf = acc*scale
//      1 -> Cf = resid + acc
//      2 -> Cb = bf16( gelu(acc + bias[n]) )
//      3 -> Cf = resid + acc + bias[n]
__global__ void __launch_bounds__(256)
gemm_bf16_wmma(const __bf16* __restrict__ A, const __bf16* __restrict__ Bt,
               int M, int N, int Kd, int epi, float scale,
               const float* __restrict__ bias, const float* __restrict__ resid,
               float* __restrict__ Cf, __bf16* __restrict__ Cb) {
    const int tid  = threadIdx.x;
    const int lane = tid & 31;
    const int w    = tid >> 5;          // 0..7
    const int wm   = w >> 2;            // 0..1  (32 rows each)
    const int wn   = w & 3;             // 0..3  (64 cols each)

    const int m0 = blockIdx.y * 64  + wm * 32;
    const int n0 = blockIdx.x * 256 + wn * 64;

    v8f acc[2][4];
    #pragma unroll
    for (int i = 0; i < 2; ++i)
        #pragma unroll
        for (int j = 0; j < 4; ++j) acc[i][j] = {};

    const int arow = (lane & 15);

    for (int k0 = 0; k0 < Kd; k0 += 32) {
        // warm L2/L0 for the K+64 lines (global_prefetch_b8)
        if (k0 + 64 < Kd) {
            __builtin_prefetch(A  + (size_t)(m0 + arow) * Kd + k0 + 64, 0, 0);
            __builtin_prefetch(Bt + (size_t)(n0 + arow) * Kd + k0 + 64, 0, 0);
        }
        bfx16 a0 = load_frag(A, m0,      Kd, k0, lane);
        bfx16 a1 = load_frag(A, m0 + 16, Kd, k0, lane);
        bfx16 b0 = load_frag(Bt, n0,      Kd, k0, lane);
        bfx16 b1 = load_frag(Bt, n0 + 16, Kd, k0, lane);
        bfx16 b2 = load_frag(Bt, n0 + 32, Kd, k0, lane);
        bfx16 b3 = load_frag(Bt, n0 + 48, Kd, k0, lane);
        acc[0][0] = __builtin_amdgcn_wmma_f32_16x16x32_bf16(false, a0, false, b0,
                        (short)0, acc[0][0], false, false);
        acc[0][1] = __builtin_amdgcn_wmma_f32_16x16x32_bf16(false, a0, false, b1,
                        (short)0, acc[0][1], false, false);
        acc[0][2] = __builtin_amdgcn_wmma_f32_16x16x32_bf16(false, a0, false, b2,
                        (short)0, acc[0][2], false, false);
        acc[0][3] = __builtin_amdgcn_wmma_f32_16x16x32_bf16(false, a0, false, b3,
                        (short)0, acc[0][3], false, false);
        acc[1][0] = __builtin_amdgcn_wmma_f32_16x16x32_bf16(false, a1, false, b0,
                        (short)0, acc[1][0], false, false);
        acc[1][1] = __builtin_amdgcn_wmma_f32_16x16x32_bf16(false, a1, false, b1,
                        (short)0, acc[1][1], false, false);
        acc[1][2] = __builtin_amdgcn_wmma_f32_16x16x32_bf16(false, a1, false, b2,
                        (short)0, acc[1][2], false, false);
        acc[1][3] = __builtin_amdgcn_wmma_f32_16x16x32_bf16(false, a1, false, b3,
                        (short)0, acc[1][3], false, false);
    }

    // C/D layout: lanes 0-15 -> M = e, lanes 16-31 -> M = 8+e ; N = lane&15
    const int col_in = lane & 15;
    const int rhalf  = (lane >> 4) * 8;
    #pragma unroll
    for (int ti = 0; ti < 2; ++ti) {
        #pragma unroll
        for (int tj = 0; tj < 4; ++tj) {
            int c  = n0 + tj * 16 + col_in;
            int mb = m0 + ti * 16 + rhalf;
            #pragma unroll
            for (int e = 0; e < 8; ++e) {
                int r = mb + e;
                float val = acc[ti][tj][e];
                size_t idx = (size_t)r * N + c;
                if (epi == 0) {
                    Cf[idx] = val * scale;
                } else if (epi == 1) {
                    Cf[idx] = resid[idx] + val;
                } else if (epi == 2) {
                    Cb[idx] = (__bf16)gelu_exact(val + bias[c]);
                } else { // 3
                    Cf[idx] = resid[idx] + val + bias[c];
                }
            }
        }
    }
}

// ---------------- weight transpose + bf16 convert -----------
// W [Kin, Nout] f32 row-major -> Wt [Nout, Kin] bf16 row-major
__global__ void wtrans_kernel(const float* __restrict__ W, __bf16* __restrict__ Wt,
                              int Kin, int Nout) {
    int idx = blockIdx.x * 256 + threadIdx.x;
    if (idx >= Kin * Nout) return;
    int n = idx / Kin;
    int k = idx - n * Kin;
    Wt[idx] = (__bf16)W[(size_t)k * Nout + n];
}

// ---------------- layernorm (row = token) -------------------
__global__ void __launch_bounds__(256)
ln_kernel(const float* __restrict__ x, const float* __restrict__ g,
          const float* __restrict__ b, __bf16* __restrict__ out) {
    __shared__ float s1[256], s2[256];
    int row = blockIdx.x, tid = threadIdx.x;
    const float* xr = x + (size_t)row * DD;
    float v0 = xr[tid], v1 = xr[tid + 256], v2 = xr[tid + 512], v3 = xr[tid + 768];
    s1[tid] = v0 + v1 + v2 + v3;
    s2[tid] = v0 * v0 + v1 * v1 + v2 * v2 + v3 * v3;
    __syncthreads();
    for (int s = 128; s > 0; s >>= 1) {
        if (tid < s) { s1[tid] += s1[tid + s]; s2[tid] += s2[tid + s]; }
        __syncthreads();
    }
    float mean = s1[0] * (1.0f / DD);
    float var  = s2[0] * (1.0f / DD) - mean * mean;
    float inv  = rsqrtf(var + EPSF);
    __bf16* orow = out + (size_t)row * DD;
    #pragma unroll
    for (int j = 0; j < 4; ++j) {
        int i = tid + j * 256;
        orow[i] = (__bf16)((xr[i] - mean) * inv * g[i] + b[i]);
    }
}

// ---------------- low-rank gate: t16 = xn @ Wgk1 ------------
__global__ void lowrank1_kernel(const __bf16* __restrict__ xn,
                                const float* __restrict__ Wgk1,
                                float* __restrict__ t16) {
    int idx = blockIdx.x * 256 + threadIdx.x;
    if (idx >= TT * 16) return;
    int t = idx >> 4, r = idx & 15;
    const __bf16* xr = xn + (size_t)t * DD;
    float acc = 0.f;
    for (int k = 0; k < DD; ++k) acc += (float)xr[k] * Wgk1[k * 16 + r];
    t16[idx] = acc;
}

// gk = logsigmoid(t16 @ Wgk2 + bgk) / 16
__global__ void gk_kernel(const float* __restrict__ t16,
                          const float* __restrict__ Wgk2,
                          const float* __restrict__ bgk,
                          float* __restrict__ gk) {
    int idx = blockIdx.x * 256 + threadIdx.x;
    if (idx >= TT * KK) return;
    int t = idx >> 9, n = idx & 511;
    float acc = bgk[n];
    const float* tr = t16 + t * 16;
    #pragma unroll
    for (int r = 0; r < 16; ++r) acc += tr[r] * Wgk2[r * KK + n];
    // stable log-sigmoid
    float ls = fminf(acc, 0.f) - log1pf(__expf(-fabsf(acc)));
    gk[idx] = ls * (1.0f / GATE_NORM);
}

// ---------------- GLA recurrence ----------------------------
// one block per (b,h); 1024 threads = (4 K-chunks) x (256 V-cols).
// Each thread keeps 32 state registers; per-step 4-way K reduction in LDS.
__global__ void __launch_bounds__(1024, 1)
gla_scan_kernel(const float* __restrict__ q, const float* __restrict__ k,
                const float* __restrict__ v, const float* __restrict__ gk,
                float* __restrict__ o) {
    int bh = blockIdx.x;
    int b = bh >> 2, h = bh & 3;
    int tid = threadIdx.x;
    int vc = tid & 255;                 // value column 0..255
    int kc = tid >> 8;                  // K chunk 0..3
    int kb = kc * 32;
    __shared__ float sq[DKk], sk[DKk], se[DKk], sv[DVv];
    __shared__ float sacc[4][DVv];
    float st[32];
    #pragma unroll
    for (int i = 0; i < 32; ++i) st[i] = 0.f;
    size_t rowbase = (size_t)b * SS;
    for (int t = 0; t < SS; ++t) {
        size_t row = rowbase + t;
        if (tid < 128) {
            sq[tid] = q[row * KK + h * DKk + tid];
        } else if (tid < 256) {
            sk[tid - 128] = k[row * KK + h * DKk + (tid - 128)];
        } else if (tid < 384) {
            se[tid - 256] = __expf(gk[row * KK + h * DKk + (tid - 256)]);
        } else if (tid < 640) {
            sv[tid - 384] = v[row * VV + h * DVv + (tid - 384)];
        }
        __syncthreads();
        float vt = sv[vc];
        float partial = 0.f;
        #pragma unroll
        for (int i = 0; i < 32; ++i) {
            st[i]   = fmaf(se[kb + i], st[i], sk[kb + i] * vt);
            partial = fmaf(sq[kb + i], st[i], partial);
        }
        sacc[kc][vc] = partial;
        __syncthreads();
        if (tid < 256) {
            o[row * VV + h * DVv + tid] =
                sacc[0][tid] + sacc[1][tid] + sacc[2][tid] + sacc[3][tid];
        }
        __syncthreads();
    }
}

// ---------------- output gate: rmsnorm(o)*g*sigmoid(g) ------
__global__ void __launch_bounds__(256)
gate_kernel(const float* __restrict__ o, const float* __restrict__ g,
            const float* __restrict__ gnw, __bf16* __restrict__ obf) {
    __shared__ float s1[256];
    int blk = blockIdx.x;              // t*H + h
    int t = blk >> 2, h = blk & 3, tid = threadIdx.x;
    size_t idx = (size_t)t * VV + h * DVv + tid;
    float val = o[idx];
    s1[tid] = val * val;
    __syncthreads();
    for (int s = 128; s > 0; s >>= 1) {
        if (tid < s) s1[tid] += s1[tid + s];
        __syncthreads();
    }
    float inv = rsqrtf(s1[0] * (1.0f / DVv) + EPSF);
    float y = val * inv * gnw[tid];
    float gv = g[idx];
    obf[idx] = (__bf16)(y * gv * sigmoidf_(gv));
}

// ============================================================
extern "C" void kernel_launch(void* const* d_in, const int* in_sizes, int n_in,
                              void* d_out, int out_size, void* d_ws, size_t ws_size,
                              hipStream_t stream) {
    const float* x      = (const float*)d_in[0];
    const float* ln1_g  = (const float*)d_in[1];
    const float* ln1_b  = (const float*)d_in[2];
    const float* Wq     = (const float*)d_in[3];
    const float* Wk     = (const float*)d_in[4];
    const float* Wv     = (const float*)d_in[5];
    const float* Wg     = (const float*)d_in[6];
    const float* Wgk1   = (const float*)d_in[7];
    const float* Wgk2   = (const float*)d_in[8];
    const float* bgk    = (const float*)d_in[9];
    const float* gnw    = (const float*)d_in[10];
    const float* Wo     = (const float*)d_in[11];
    const float* ln2_g  = (const float*)d_in[12];
    const float* ln2_b  = (const float*)d_in[13];
    const float* W1     = (const float*)d_in[14];
    const float* b1     = (const float*)d_in[15];
    const float* W2     = (const float*)d_in[16];
    const float* b2     = (const float*)d_in[17];
    float* out = (float*)d_out;

    // ---- workspace carving (256B aligned) ----
    size_t off = 0;
    auto carve = [&](size_t bytes) -> char* {
        char* p = (char*)d_ws + off;
        off = (off + bytes + 255) & ~(size_t)255;
        return p;
    };
    __bf16* wq_t = (__bf16*)carve((size_t)KK * DD * 2);
    __bf16* wk_t = (__bf16*)carve((size_t)KK * DD * 2);
    __bf16* wv_t = (__bf16*)carve((size_t)VV * DD * 2);
    __bf16* wg_t = (__bf16*)carve((size_t)VV * DD * 2);
    __bf16* wo_t = (__bf16*)carve((size_t)DD * VV * 2);
    __bf16* w1_t = (__bf16*)carve((size_t)MLPH * DD * 2);
    __bf16* w2_t = (__bf16*)carve((size_t)DD * MLPH * 2);
    __bf16* xn   = (__bf16*)carve((size_t)TT * DD * 2);
    float*  qb   = (float*)carve((size_t)TT * KK * 4);
    float*  kb   = (float*)carve((size_t)TT * KK * 4);
    float*  vb   = (float*)carve((size_t)TT * VV * 4);
    float*  gb   = (float*)carve((size_t)TT * VV * 4);
    float*  t16  = (float*)carve((size_t)TT * 16 * 4);
    float*  gkb  = (float*)carve((size_t)TT * KK * 4);
    float*  ob   = (float*)carve((size_t)TT * VV * 4);
    __bf16* obf  = (__bf16*)carve((size_t)TT * VV * 2);
    float*  x2   = (float*)carve((size_t)TT * DD * 4);
    __bf16* hbf  = (__bf16*)carve((size_t)TT * DD * 2);
    __bf16* m1bf = (__bf16*)carve((size_t)TT * MLPH * 2);
    (void)ws_size;

    // ---- 1. weight convert + transpose (f32 -> bf16 [N,K]) ----
    auto wt = [&](const float* W, __bf16* Wt, int Kin, int Nout) {
        int n = Kin * Nout;
        wtrans_kernel<<<(n + 255) / 256, 256, 0, stream>>>(W, Wt, Kin, Nout);
    };
    wt(Wq, wq_t, DD, KK);
    wt(Wk, wk_t, DD, KK);
    wt(Wv, wv_t, DD, VV);
    wt(Wg, wg_t, DD, VV);
    wt(Wo, wo_t, VV, DD);
    wt(W1, w1_t, DD, MLPH);
    wt(W2, w2_t, MLPH, DD);

    // ---- 2. LN1 ----
    ln_kernel<<<TT, 256, 0, stream>>>(x, ln1_g, ln1_b, xn);

    // ---- 3..6. projections via WMMA GEMM ----
    const float qscale = 0.08838834764831845f; // 128^-0.5
    dim3 blk(256);
    {
        dim3 grid(KK / 256, TT / 64);
        gemm_bf16_wmma<<<grid, blk, 0, stream>>>(xn, wq_t, TT, KK, DD, 0, qscale,
                                                 nullptr, nullptr, qb, nullptr);
        gemm_bf16_wmma<<<grid, blk, 0, stream>>>(xn, wk_t, TT, KK, DD, 0, 1.0f,
                                                 nullptr, nullptr, kb, nullptr);
    }
    {
        dim3 grid(VV / 256, TT / 64);
        gemm_bf16_wmma<<<grid, blk, 0, stream>>>(xn, wv_t, TT, VV, DD, 0, 1.0f,
                                                 nullptr, nullptr, vb, nullptr);
        gemm_bf16_wmma<<<grid, blk, 0, stream>>>(xn, wg_t, TT, VV, DD, 0, 1.0f,
                                                 nullptr, nullptr, gb, nullptr);
    }

    // ---- 7..8. low-rank gate path ----
    lowrank1_kernel<<<(TT * 16 + 255) / 256, 256, 0, stream>>>(xn, Wgk1, t16);
    gk_kernel<<<(TT * KK + 255) / 256, 256, 0, stream>>>(t16, Wgk2, bgk, gkb);

    // ---- 9. GLA recurrence (1 WGP-full workgroup per (b,h)) ----
    gla_scan_kernel<<<BB * HH, 1024, 0, stream>>>(qb, kb, vb, gkb, ob);

    // ---- 10. output gate ----
    gate_kernel<<<TT * HH, 256, 0, stream>>>(ob, gb, gnw, obf);

    // ---- 11. Wo GEMM + residual ----
    {
        dim3 grid(DD / 256, TT / 64);
        gemm_bf16_wmma<<<grid, blk, 0, stream>>>(obf, wo_t, TT, DD, VV, 1, 1.0f,
                                                 nullptr, x, x2, nullptr);
    }

    // ---- 12. LN2 ----
    ln_kernel<<<TT, 256, 0, stream>>>(x2, ln2_g, ln2_b, hbf);

    // ---- 13. MLP up (bias + exact GELU, bf16 out) ----
    {
        dim3 grid(MLPH / 256, TT / 64);
        gemm_bf16_wmma<<<grid, blk, 0, stream>>>(hbf, w1_t, TT, MLPH, DD, 2, 1.0f,
                                                 b1, nullptr, nullptr, m1bf);
    }

    // ---- 14. MLP down (bias + residual, f32 out) ----
    {
        dim3 grid(DD / 256, TT / 64);
        gemm_bf16_wmma<<<grid, blk, 0, stream>>>(m1bf, w2_t, TT, DD, MLPH, 3, 1.0f,
                                                 b2, x2, out, nullptr);
    }
}
